// Attention_90787018703480
// MI455X (gfx1250) — matrix operand order repaired and checked by reading, not compiled, not run on hardware
//
#include <hip/hip_runtime.h>

#define D_MODEL 2048
#define SEQ_T   2048
#define NHEADS  16
#define HEAD_D  128
#define BATCH   4
#define M_TOTAL (BATCH*SEQ_T)   // 8192

typedef __attribute__((ext_vector_type(16))) __bf16 v16bf;
typedef __attribute__((ext_vector_type(8)))  __bf16 v8bf;
typedef __attribute__((ext_vector_type(4)))  __bf16 v4bf;
typedef __attribute__((ext_vector_type(8)))  float  v8f;

static __device__ __forceinline__ v8f wmma_bf16(v16bf a, v16bf b, v8f c) {
  return __builtin_amdgcn_wmma_f32_16x16x32_bf16(
      /*neg_a=*/false, a, /*neg_b=*/false, b,
      /*c_mod=*/(short)0, c, /*reuse_a=*/false, /*reuse_b=*/false);
}

static __device__ __forceinline__ v16bf cat8(v8bf lo, v8bf hi) {
  return __builtin_shufflevector(lo, hi, 0,1,2,3,4,5,6,7,8,9,10,11,12,13,14,15);
}

// async 16B global -> LDS copy (ASYNCcnt-tracked, no VGPR data path)
static __device__ __forceinline__ void async_b128(const __bf16* g, __bf16* l) {
  unsigned lds = (unsigned)(unsigned long long)l;          // low 32 bits = LDS offset
  unsigned long long ga = (unsigned long long)g;
  asm volatile("global_load_async_to_lds_b128 %0, %1, off"
               :: "v"(lds), "v"(ga) : "memory");
}

// ---------------- f32 -> bf16 conversion (4-wide) ----------------
__global__ __launch_bounds__(256) void k_cvt_bf16(const float* __restrict__ s,
                                                  __bf16* __restrict__ d, int n) {
  int i = (blockIdx.x * 256 + threadIdx.x) * 4;
  if (i < n) {
    float4 f = *(const float4*)(s + i);
    v4bf p;
    p[0] = (__bf16)f.x; p[1] = (__bf16)f.y; p[2] = (__bf16)f.z; p[3] = (__bf16)f.w;
    *(v4bf*)(d + i) = p;
  }
}

// ---------------- GEMM: C[m,n] = sum_k A[m,k] * W[n,k]  (y = A @ W^T) -------
// Block tile 128x128, K-step 64, double-buffered async LDS staging (64KB).
// Wave w computes rows [w*16, w*16+16) x all 128 cols (8 accumulators).
// mode 0: Q -> RoPE + 1/sqrt(Dh), store bf16 (B,H,T,Dh)
// mode 1: K -> RoPE,              store bf16 (B,H,T,Dh)
// mode 2: V ->                    store bf16 transposed (B,H,Dh,T)
// mode 3: O ->                    store f32 row-major (M, D)
__global__ __launch_bounds__(256) void k_gemm_bf16(const __bf16* __restrict__ A,
                                                   const __bf16* __restrict__ W,
                                                   __bf16* __restrict__ obf,
                                                   float* __restrict__ of32,
                                                   int mode) {
  __shared__ __bf16 sA[2 * 128 * 64];   // 2 x 16KB
  __shared__ __bf16 sB[2 * 128 * 64];   // 2 x 16KB

  const int tid  = threadIdx.x;
  const int lane = tid & 31;
  const int wave = tid >> 5;
  const int l15  = lane & 15;
  const int hi   = lane >> 4;
  const int mblk = blockIdx.y * 128;
  const int n0   = blockIdx.x * 128;
  const int m0   = mblk + wave * 16;

  const __bf16* Ag = A + (size_t)mblk * D_MODEL;
  const __bf16* Wg = W + (size_t)n0 * D_MODEL;

  // stage: 128x64 bf16 tile = 1024 x 16B chunks; 256 threads x 4 chunks each
  auto stage = [&](int buf, int k0) {
#pragma unroll
    for (int t = 0; t < 4; ++t) {
      int c   = tid + t * 256;
      int row = c >> 3;                   // 8 chunks per 128B row
      int off = (c & 7) * 8;              // elements
      async_b128(Ag + (size_t)row * D_MODEL + k0 + off, sA + buf * 8192 + row * 64 + off);
      async_b128(Wg + (size_t)row * D_MODEL + k0 + off, sB + buf * 8192 + row * 64 + off);
    }
  };

  v8f acc[8] = {};

  stage(0, 0);
  for (int k0 = 0; k0 < D_MODEL; k0 += 64) {
    int cur = (k0 >> 6) & 1;
    if (k0 + 64 < D_MODEL) {
      stage(cur ^ 1, k0 + 64);
      asm volatile("s_wait_asynccnt 0x8" ::: "memory"); // current buffer landed
    } else {
      asm volatile("s_wait_asynccnt 0x0" ::: "memory");
    }
    __syncthreads();

    const __bf16* la = sA + cur * 8192;
    const __bf16* lb = sB + cur * 8192;
    const __bf16* ar = la + (wave * 16 + l15) * 64;
    v16bf af0 = cat8(*(const v8bf*)(ar + hi * 16),      *(const v8bf*)(ar + 8 + hi * 16));
    v16bf af1 = cat8(*(const v8bf*)(ar + 32 + hi * 16), *(const v8bf*)(ar + 40 + hi * 16));

#pragma unroll
    for (int half = 0; half < 2; ++half) {
      v16bf bfr[8];
#pragma unroll
      for (int a = 0; a < 8; ++a)
        bfr[a] = *(const v16bf*)(lb + (a * 16 + l15) * 64 + half * 32 + hi * 16);
      v16bf af = half ? af1 : af0;
#pragma unroll
      for (int a = 0; a < 8; ++a)
        acc[a] = wmma_bf16(af, bfr[a], acc[a]);
    }
    __syncthreads();   // all waves done reading before buffer is overwritten
  }

  // ---- epilogue ----
  const int b     = m0 / SEQ_T;
  const int tbase = (m0 % SEQ_T) + hi * 8;

  if (mode <= 1) {
    const float qscale = (mode == 0) ? 0.08838834764831845f : 1.0f; // 1/sqrt(128)
#pragma unroll
    for (int a = 0; a < 4; ++a) {
      int n_lo = n0 + a * 16 + l15;
      int h = n_lo >> 7;
      int d = n_lo & 127;                 // < 64 since a < 4
      float theta = __expf(-(float)d * (9.210340371976184f / 64.0f)); // 10000^(-d/64)
      __bf16* qb = obf + (size_t)(b * NHEADS + h) * SEQ_T * HEAD_D;
#pragma unroll
      for (int r = 0; r < 8; ++r) {
        int t = tbase + r;
        float sv, cv;
        __sincosf((float)t * theta, &sv, &cv);
        float lo = acc[a][r], hv = acc[a + 4][r];
        float olo = (lo * cv - hv * sv) * qscale;
        float ohi = (hv * cv + lo * sv) * qscale;
        size_t roff = (size_t)t * HEAD_D;
        qb[roff + d]      = (__bf16)olo;
        qb[roff + d + 64] = (__bf16)ohi;
      }
    }
  } else if (mode == 2) {
#pragma unroll
    for (int a = 0; a < 8; ++a) {
      int n = n0 + a * 16 + l15;
      int h = n >> 7, d = n & 127;
      __bf16* vb = obf + ((size_t)(b * NHEADS + h) * HEAD_D + d) * SEQ_T;
#pragma unroll
      for (int r = 0; r < 8; ++r)
        vb[tbase + r] = (__bf16)acc[a][r];
    }
  } else {
#pragma unroll
    for (int a = 0; a < 8; ++a) {
      int n = n0 + a * 16 + l15;
#pragma unroll
      for (int r = 0; r < 8; ++r)
        of32[(size_t)(m0 + hi * 8 + r) * D_MODEL + n] = acc[a][r];
    }
  }
}

// ---------------- Flash attention: one wave per 16-query tile ----------------
__global__ __launch_bounds__(256) void k_attn(const __bf16* __restrict__ Q,
                                              const __bf16* __restrict__ K,
                                              const __bf16* __restrict__ Vt,
                                              __bf16* __restrict__ Ctx) {
  __shared__ __bf16 sP[8 * 512];              // per-wave 16x32 bf16 P tile (col-major)
  const int lane = threadIdx.x & 31;
  const int wave = threadIdx.x >> 5;
  const int l15  = lane & 15;
  const int hi   = lane >> 4;
  const int bh   = blockIdx.y;
  const int b    = bh >> 4, h = bh & 15;
  const int qt0  = (blockIdx.x * 8 + wave) * 16;

  const __bf16* Qb = Q  + (size_t)bh * SEQ_T * HEAD_D;
  const __bf16* Kb = K  + (size_t)bh * SEQ_T * HEAD_D;
  const __bf16* Vb = Vt + (size_t)bh * HEAD_D * SEQ_T;
  __bf16* sp = sP + wave * 512;

  // Q A-fragments for the 4 K-chunks over Dh=128 (resident)
  v16bf qa[4];
  const __bf16* qrow = Qb + (size_t)(qt0 + l15) * HEAD_D;
#pragma unroll
  for (int c = 0; c < 4; ++c) {
    int dk = c * 32;
    v8bf lo = *(const v8bf*)(qrow + dk + hi * 16);
    v8bf hh = *(const v8bf*)(qrow + dk + 8 + hi * 16);
    qa[c] = cat8(lo, hh);
  }

  v8f oacc[8] = {};
  float mrow[8], lrow[8];
#pragma unroll
  for (int r = 0; r < 8; ++r) { mrow[r] = -1e30f; lrow[r] = 0.0f; }

  for (int kt = 0; kt < SEQ_T; kt += 32) {
    // batch-load all 8 K fragments, then 8 score WMMAs
    v16bf kf[8];
#pragma unroll
    for (int c = 0; c < 4; ++c) {
      kf[c]     = *(const v16bf*)(Kb + (size_t)(kt + l15) * HEAD_D + c * 32 + hi * 16);
      kf[4 + c] = *(const v16bf*)(Kb + (size_t)(kt + 16 + l15) * HEAD_D + c * 32 + hi * 16);
    }
    v8f s0 = {}, s1 = {};
#pragma unroll
    for (int c = 0; c < 4; ++c) {
      s0 = wmma_bf16(qa[c], kf[c], s0);
      s1 = wmma_bf16(qa[c], kf[4 + c], s1);
    }

    // issue V fragment loads now; latency hidden under softmax + LDS transpose
    v16bf vf[8];
#pragma unroll
    for (int a = 0; a < 8; ++a)
      vf[a] = *(const v16bf*)(Vb + (size_t)(a * 16 + l15) * SEQ_T + kt + hi * 16);

    // online softmax (rows = VGPR slots, cols = lanes; reduce per 16-lane half)
    float p0[8], p1[8];
#pragma unroll
    for (int r = 0; r < 8; ++r) {
      float mx = fmaxf(s0[r], s1[r]);
      mx = fmaxf(mx, __shfl_xor(mx, 1));
      mx = fmaxf(mx, __shfl_xor(mx, 2));
      mx = fmaxf(mx, __shfl_xor(mx, 4));
      mx = fmaxf(mx, __shfl_xor(mx, 8));
      float mn    = fmaxf(mrow[r], mx);
      float alpha = __expf(mrow[r] - mn);
      mrow[r] = mn;
      float e0 = __expf(s0[r] - mn);
      float e1 = __expf(s1[r] - mn);
      p0[r] = e0; p1[r] = e1;
      float rs = e0 + e1;
      rs += __shfl_xor(rs, 1);
      rs += __shfl_xor(rs, 2);
      rs += __shfl_xor(rs, 4);
      rs += __shfl_xor(rs, 8);
      lrow[r] = lrow[r] * alpha + rs;
#pragma unroll
      for (int a = 0; a < 8; ++a) oacc[a][r] = oacc[a][r] * alpha;
    }

    // P (D-layout) -> LDS column-major [kk][m] -> A-layout fragment
    v8bf pk0, pk1;
#pragma unroll
    for (int r = 0; r < 8; ++r) { pk0[r] = (__bf16)p0[r]; pk1[r] = (__bf16)p1[r]; }
    int colbase = l15 * 16 + hi * 8;
    *(v8bf*)(sp + colbase)       = pk0;       // kk = l15
    *(v8bf*)(sp + 256 + colbase) = pk1;       // kk = 16 + l15
    asm volatile("s_wait_dscnt 0x0" ::: "memory");

    v16bf pa;
    int kb = hi * 16;
#pragma unroll
    for (int i = 0; i < 8; ++i) pa[i]     = sp[(kb + i) * 16 + l15];
#pragma unroll
    for (int i = 0; i < 8; ++i) pa[8 + i] = sp[(kb + 8 + i) * 16 + l15];

    // out += P @ V
#pragma unroll
    for (int a = 0; a < 8; ++a)
      oacc[a] = wmma_bf16(pa, vf[a], oacc[a]);
  }

  // normalize + store context bf16 as (B,T,D) row-major
  int tq = qt0 + hi * 8;
#pragma unroll
  for (int a = 0; a < 8; ++a) {
    int d = a * 16 + l15;
#pragma unroll
    for (int r = 0; r < 8; ++r) {
      float v = oacc[a][r] / lrow[r];
      Ctx[(size_t)(b * SEQ_T + tq + r) * D_MODEL + h * HEAD_D + d] = (__bf16)v;
    }
  }
}

extern "C" void kernel_launch(void* const* d_in, const int* in_sizes, int n_in,
                              void* d_out, int out_size, void* d_ws, size_t ws_size,
                              hipStream_t stream) {
  const float* x  = (const float*)d_in[0];
  const float* Wq = (const float*)d_in[1];
  const float* Wk = (const float*)d_in[2];
  const float* Wv = (const float*)d_in[3];
  const float* Wo = (const float*)d_in[4];

  const size_t NELEM_X = (size_t)M_TOTAL * D_MODEL;   // 16,777,216
  const size_t NELEM_W = (size_t)D_MODEL * D_MODEL;   //  4,194,304

  char* w = (char*)d_ws;
  __bf16* Xbf = (__bf16*)w;  w += NELEM_X * 2;
  __bf16* Wqb = (__bf16*)w;  w += NELEM_W * 2;
  __bf16* Wkb = (__bf16*)w;  w += NELEM_W * 2;
  __bf16* Wvb = (__bf16*)w;  w += NELEM_W * 2;
  __bf16* Wob = (__bf16*)w;  w += NELEM_W * 2;
  __bf16* Qb  = (__bf16*)w;  w += NELEM_X * 2;
  __bf16* Kb  = (__bf16*)w;  w += NELEM_X * 2;
  __bf16* Vtb = (__bf16*)w;  w += NELEM_X * 2;
  __bf16* Ctx = (__bf16*)w;  w += NELEM_X * 2;

  k_cvt_bf16<<<(int)(NELEM_X / 4 + 255) / 256, 256, 0, stream>>>(x,  Xbf, (int)NELEM_X);
  k_cvt_bf16<<<(int)(NELEM_W / 4 + 255) / 256, 256, 0, stream>>>(Wq, Wqb, (int)NELEM_W);
  k_cvt_bf16<<<(int)(NELEM_W / 4 + 255) / 256, 256, 0, stream>>>(Wk, Wkb, (int)NELEM_W);
  k_cvt_bf16<<<(int)(NELEM_W / 4 + 255) / 256, 256, 0, stream>>>(Wv, Wvb, (int)NELEM_W);
  k_cvt_bf16<<<(int)(NELEM_W / 4 + 255) / 256, 256, 0, stream>>>(Wo, Wob, (int)NELEM_W);

  dim3 gg(D_MODEL / 128, M_TOTAL / 128);   // (16, 64)
  k_gemm_bf16<<<gg, 256, 0, stream>>>(Xbf, Wqb, Qb,  nullptr, 0);
  k_gemm_bf16<<<gg, 256, 0, stream>>>(Xbf, Wkb, Kb,  nullptr, 1);
  k_gemm_bf16<<<gg, 256, 0, stream>>>(Xbf, Wvb, Vtb, nullptr, 2);

  dim3 ga(SEQ_T / 16 / 8, BATCH * NHEADS); // (16, 64)
  k_attn<<<ga, 256, 0, stream>>>(Qb, Kb, Vtb, Ctx);

  k_gemm_bf16<<<gg, 256, 0, stream>>>(Ctx, Wob, nullptr, (float*)d_out, 3);
}
